// LSTM_34583076667524
// MI455X (gfx1250) — matrix operand order repaired and checked by reading, not compiled, not run on hardware
//
#include <hip/hip_runtime.h>

// ---------------------------------------------------------------------------
// LSTM for MI455X (gfx1250): bf16 WMMA GEMMs, f32 accumulate + f32 gate math.
// Phase 1: convert X / pack W,b to bf16 workspace.
// Phase 2: xg = X @ Wx^T + bias  (WMMA GEMM, double-buffered LDS, stored bf16).
// Phase 3: 512 sequential fused step kernels: whole h_{t-1} (128KB) pulled
//          into LDS by one TDM tensor_load_to_lds DMA, then a barrier-free
//          K=1024 WMMA loop; gates column-partitioned per workgroup so the
//          sigmoid/tanh/state update happens in the same launch.
// ---------------------------------------------------------------------------

typedef __bf16 bf16_t;
typedef __attribute__((ext_vector_type(16))) __bf16 v16bf;
typedef __attribute__((ext_vector_type(8)))  __bf16 v8bf;
typedef __attribute__((ext_vector_type(8)))  float  v8f;
typedef __attribute__((ext_vector_type(4)))  unsigned int u32x4;
typedef __attribute__((ext_vector_type(8)))  int i32x8;
typedef __attribute__((ext_vector_type(4)))  int i32x4;

union BFfrag { v16bf v; v8bf h[2]; };

#define T_STEPS 512
#define BATCH   64
#define EDIM    1024
#define HDIM    1024
#define G4H     4096   // 4*HDIM

// ---------------------------------------------------------------------------
// Phase 1a: pack four [H, E+H] fp32 gate matrices -> bf16 Wx [4H,E], Wh [4H,H].
// ---------------------------------------------------------------------------
__global__ void pack_weights_kernel(const float* __restrict__ Wf,
                                    const float* __restrict__ Wi,
                                    const float* __restrict__ Wo,
                                    const float* __restrict__ Wc,
                                    bf16_t* __restrict__ Wx,
                                    bf16_t* __restrict__ Wh) {
    size_t idx = (size_t)blockIdx.x * blockDim.x + threadIdx.x; // < 4*1024*2048
    int g   = (int)(idx >> 21);
    int rem = (int)(idx & ((1u << 21) - 1)); // j*2048 + col
    int j   = rem >> 11;
    int col = rem & 2047;
    const float* W = (g == 0) ? Wf : (g == 1) ? Wi : (g == 2) ? Wo : Wc;
    float v = W[rem];
    if (col < EDIM)
        Wx[((size_t)(g * HDIM + j)) * EDIM + col] = (bf16_t)v;
    else
        Wh[((size_t)(g * HDIM + j)) * HDIM + (col - EDIM)] = (bf16_t)v;
}

// ---------------------------------------------------------------------------
// Phase 1b: embedding fp32 -> bf16 (x4 vectorized).
// ---------------------------------------------------------------------------
__global__ void convert_x_kernel(const float* __restrict__ X,
                                 bf16_t* __restrict__ Xb) {
    size_t i = ((size_t)blockIdx.x * blockDim.x + threadIdx.x) * 4;
    float4 v = *(const float4*)(X + i);
    Xb[i + 0] = (bf16_t)v.x;
    Xb[i + 1] = (bf16_t)v.y;
    Xb[i + 2] = (bf16_t)v.z;
    Xb[i + 3] = (bf16_t)v.w;
}

// ---------------------------------------------------------------------------
// Phase 1c: concat bias [4H], zero h ping-pong buffers and cell state.
// ---------------------------------------------------------------------------
__global__ void init_state_bias_kernel(const float* __restrict__ bf_,
                                       const float* __restrict__ bi_,
                                       const float* __restrict__ bo_,
                                       const float* __restrict__ bc_,
                                       float* __restrict__ bias,
                                       bf16_t* __restrict__ h0,
                                       bf16_t* __restrict__ h1,
                                       float* __restrict__ c) {
    int i = blockIdx.x * blockDim.x + threadIdx.x; // 0..65535
    if (i < BATCH * HDIM) {
        h0[i] = (bf16_t)0.0f;
        h1[i] = (bf16_t)0.0f;
        c[i]  = 0.0f;
    }
    if (i < G4H) {
        int g = i >> 10, j = i & 1023;
        const float* src = (g == 0) ? bf_ : (g == 1) ? bi_ : (g == 2) ? bo_ : bc_;
        bias[i] = src[j];
    }
}

// ---------------------------------------------------------------------------
// Phase 2: xg[M=T*B, N=4H] = Xb[M,K=E] @ Wx[N,K]^T + bias, stored bf16.
// 256 threads (8 waves), 128x128 block tile; wave = 32(M) x 64(N);
// K-step 32, double-buffered LDS (one barrier per K-iter).
// ---------------------------------------------------------------------------
__global__ __launch_bounds__(256)
void xg_gemm_kernel(const bf16_t* __restrict__ Xb,
                    const bf16_t* __restrict__ Wx,
                    const float* __restrict__ bias,
                    bf16_t* __restrict__ xg) {
    __shared__ __attribute__((aligned(16))) bf16_t As[2][128][32];
    __shared__ __attribute__((aligned(16))) bf16_t Bs[2][128][32];

    const int tid  = threadIdx.x;
    const int lane = tid & 31;
    const int l16  = lane & 15;
    const int hi   = lane >> 4;
    const int wid  = tid >> 5;
    const int mw   = wid & 3;       // M wave slot (x32)
    const int nw   = wid >> 2;      // N wave slot (x64)

    const int mBase = blockIdx.x * 128;
    const int nBase = blockIdx.y * 128;

    v8f acc[2][4] = {};

    auto stage = [&](int buf, int k0) {
        #pragma unroll
        for (int i = 0; i < 2; ++i) {
            int q    = tid + i * 256;      // chunk 0..511
            int row  = q >> 2;
            int col0 = (q & 3) * 8;
            *(v8bf*)&As[buf][row][col0] =
                *(const v8bf*)&Xb[(size_t)(mBase + row) * EDIM + k0 + col0];
            *(v8bf*)&Bs[buf][row][col0] =
                *(const v8bf*)&Wx[(size_t)(nBase + row) * EDIM + k0 + col0];
        }
    };

    stage(0, 0);
    __syncthreads();

    for (int k0 = 0; k0 < EDIM; k0 += 32) {
        const int buf = (k0 >> 5) & 1;
        if (k0 + 32 < EDIM) stage(buf ^ 1, k0 + 32);

        BFfrag af[2], bfr[4];
        #pragma unroll
        for (int mt = 0; mt < 2; ++mt) {
            int row = mw * 32 + mt * 16 + l16;
            af[mt].h[0] = *(const v8bf*)&As[buf][row][hi * 8];
            af[mt].h[1] = *(const v8bf*)&As[buf][row][16 + hi * 8];
        }
        #pragma unroll
        for (int nt = 0; nt < 4; ++nt) {
            int col = nw * 64 + nt * 16 + l16;
            bfr[nt].h[0] = *(const v8bf*)&Bs[buf][col][hi * 8];
            bfr[nt].h[1] = *(const v8bf*)&Bs[buf][col][16 + hi * 8];
        }
        #pragma unroll
        for (int mt = 0; mt < 2; ++mt)
            #pragma unroll
            for (int nt = 0; nt < 4; ++nt)
                acc[mt][nt] = __builtin_amdgcn_wmma_f32_16x16x32_bf16(
                    false, af[mt].v, false, bfr[nt].v,
                    (short)0, acc[mt][nt], false, false);
        __syncthreads();
    }

    #pragma unroll
    for (int mt = 0; mt < 2; ++mt) {
        #pragma unroll
        for (int nt = 0; nt < 4; ++nt) {
            int gn  = nBase + nw * 64 + nt * 16 + l16;
            float b = bias[gn];
            #pragma unroll
            for (int r = 0; r < 8; ++r) {
                int gm = mBase + mw * 32 + mt * 16 + hi * 8 + r;
                xg[(size_t)gm * G4H + gn] = (bf16_t)(acc[mt][nt][r] + b);
            }
        }
    }
}

// ---------------------------------------------------------------------------
// Phase 3: fused LSTM step. Whole h_{t-1} [64,1024] bf16 (128KB) DMA'd to
// dynamic LDS by the Tensor Data Mover once; barrier-free K loop after that.
// 32 workgroups; wg w owns cols [w*32, w*32+32) of all FOUR gates.
// 8 waves: gate = wid>>1, 16-col subtile = wid&1, M=64 in 4 tiles.
// ---------------------------------------------------------------------------
__global__ __launch_bounds__(256)
void lstm_step_kernel(const bf16_t* __restrict__ xg,
                      const bf16_t* __restrict__ Wh,
                      const bf16_t* __restrict__ h_in,
                      bf16_t* __restrict__ h_out,
                      float* __restrict__ c,
                      float* __restrict__ out,
                      int t) {
    __shared__ float Gs[4][64][32];          // static 32KB: gate exchange
    extern __shared__ bf16_t Ah[];           // dynamic 128KB: h tile [64][1024]

    const int tid  = threadIdx.x;
    const int lane = tid & 31;
    const int l16  = lane & 15;
    const int hi   = lane >> 4;
    const int wid  = tid >> 5;
    const int gate = wid >> 1;
    const int jt   = wid & 1;
    const int j0   = blockIdx.x * 32 + jt * 16;
    const int brow = gate * HDIM + j0 + l16;   // Wh row (= N column) for lane

#if defined(__HIP_DEVICE_COMPILE__) && __has_builtin(__builtin_amdgcn_tensor_load_to_lds)
    // --- TDM DMA: whole h_in (2D tensor, dim0=1024 elems of 2B, dim1=64) ---
    if (wid == 0) {
        unsigned long ga = (unsigned long)h_in;
        u32x4 g0;
        g0[0] = 0x1u;                                     // count=1, user mode
        g0[1] = __builtin_amdgcn_groupstaticsize();       // lds_addr (dyn LDS base)
        g0[2] = (unsigned)ga;                             // global_addr[31:0]
        g0[3] = (unsigned)((ga >> 32) & 0x1FFFFFFu)       // global_addr[56:32]
              | (2u << 30);                               // type=2 ("image")
        i32x8 g1;
        g1[0] = (int)(1u << 16);          // workgroup_mask=0, data_size=1 (2B)
        g1[1] = (int)(1024u << 16);       // tensor_dim0=1024 (low16 @ [31:16])
        g1[2] = (int)(64u << 16);         // dim0 hi=0 | tensor_dim1=64 low16
        g1[3] = (int)(1024u << 16);       // dim1 hi=0 | tile_dim0=1024
        g1[4] = 64;                       // tile_dim1=64 | tile_dim2=0
        g1[5] = 1024;                     // tensor_dim0_stride low32
        g1[6] = 0;                        // stride0 hi | stride1 low
        g1[7] = 0;                        // stride1 hi
        i32x4 z4 = {};
#if __has_include(<hip/amd_detail/amd_gfx1250_TDM.h>)
        i32x8 z8 = {};
        __builtin_amdgcn_tensor_load_to_lds(g0, g1, z4, z4, z8, 0);   // clang-23 form
#else
        __builtin_amdgcn_tensor_load_to_lds(g0, g1, z4, z4, 0);       // ROCm 7.2 form
#endif
        __builtin_amdgcn_s_wait_tensorcnt(0);
    }
#else
    // Fallback: cooperative load, 16B per thread per pass.
    #pragma unroll
    for (int p = 0; p < 32; ++p) {
        int q = tid + p * 256;            // 8192 chunks of 8 bf16
        *(v8bf*)&Ah[q * 8] = *(const v8bf*)&h_in[q * 8];
    }
#endif
    __syncthreads();

    v8f acc[4] = {};

    for (int k0 = 0; k0 < HDIM; k0 += 32) {
        BFfrag bfr;
        bfr.h[0] = *(const v8bf*)&Wh[(size_t)brow * HDIM + k0 + hi * 8];
        bfr.h[1] = *(const v8bf*)&Wh[(size_t)brow * HDIM + k0 + 16 + hi * 8];

        #pragma unroll
        for (int mt = 0; mt < 4; ++mt) {
            int row = mt * 16 + l16;
            BFfrag af;
            af.h[0] = *(const v8bf*)&Ah[row * HDIM + k0 + hi * 8];
            af.h[1] = *(const v8bf*)&Ah[row * HDIM + k0 + 16 + hi * 8];
            acc[mt] = __builtin_amdgcn_wmma_f32_16x16x32_bf16(
                false, af.v, false, bfr.v, (short)0, acc[mt], false, false);
        }
    }

    // exchange gate partials through LDS
    #pragma unroll
    for (int mt = 0; mt < 4; ++mt)
        #pragma unroll
        for (int r = 0; r < 8; ++r)
            Gs[gate][mt * 16 + hi * 8 + r][jt * 16 + l16] = acc[mt][r];
    __syncthreads();

    // gate math + state update: 2048 elements, 8 per thread
    #pragma unroll
    for (int i = 0; i < 8; ++i) {
        int idx = tid + i * 256;
        int b   = idx >> 5;
        int jj  = idx & 31;
        int j   = blockIdx.x * 32 + jj;
        size_t xb = ((size_t)t * BATCH + b) * G4H + j;
        float gf = Gs[0][b][jj] + (float)xg[xb];
        float gi = Gs[1][b][jj] + (float)xg[xb + HDIM];
        float go = Gs[2][b][jj] + (float)xg[xb + 2 * HDIM];
        float gc = Gs[3][b][jj] + (float)xg[xb + 3 * HDIM];
        float f  = 1.0f / (1.0f + __expf(-gf));
        float ig = 1.0f / (1.0f + __expf(-gi));
        float o  = 1.0f / (1.0f + __expf(-go));
        float ct = tanhf(gc);
        int ci   = b * HDIM + j;
        float cn = f * c[ci] + ig * ct;
        float hn = o * tanhf(cn);
        c[ci] = cn;
        h_out[ci] = (bf16_t)hn;
        out[((size_t)b * T_STEPS + t) * HDIM + j] = hn;
    }
}

// ---------------------------------------------------------------------------
extern "C" void kernel_launch(void* const* d_in, const int* in_sizes, int n_in,
                              void* d_out, int out_size, void* d_ws, size_t ws_size,
                              hipStream_t stream) {
    (void)in_sizes; (void)n_in; (void)out_size; (void)ws_size;

    const float* X   = (const float*)d_in[0];
    const float* Wf  = (const float*)d_in[1];
    const float* bfv = (const float*)d_in[2];
    const float* Wi  = (const float*)d_in[3];
    const float* biv = (const float*)d_in[4];
    const float* Wo  = (const float*)d_in[5];
    const float* bov = (const float*)d_in[6];
    const float* Wc  = (const float*)d_in[7];
    const float* bcv = (const float*)d_in[8];
    float* out = (float*)d_out;

    char* ws = (char*)d_ws;
    const size_t MB = 1024u * 1024u;
    bf16_t* WxB  = (bf16_t*)(ws);                 //   8 MB: Wx bf16 [4H,E]
    bf16_t* WhB  = (bf16_t*)(ws + 8 * MB);        //   8 MB: Wh bf16 [4H,H]
    bf16_t* Xb   = (bf16_t*)(ws + 16 * MB);       //  64 MB: X bf16 [T*B,E]
    bf16_t* xg   = (bf16_t*)(ws + 80 * MB);       // 256 MB: xg bf16 [T*B,4H]
    float*  bias = (float*)(ws + 336 * MB);       //  16 KB
    bf16_t* h0   = (bf16_t*)(ws + 337 * MB);      // 128 KB
    bf16_t* h1   = (bf16_t*)(ws + 338 * MB);      // 128 KB
    float*  cbuf = (float*)(ws + 339 * MB);       // 256 KB

    // Phase 1
    pack_weights_kernel<<<(4 * 1024 * 2048) / 256, 256, 0, stream>>>(
        Wf, Wi, Wo, Wc, WxB, WhB);
    convert_x_kernel<<<(T_STEPS * BATCH * EDIM) / (4 * 256), 256, 0, stream>>>(X, Xb);
    init_state_bias_kernel<<<(BATCH * HDIM) / 256, 256, 0, stream>>>(
        bfv, biv, bov, bcv, bias, h0, h1, cbuf);

    // Phase 2: input projection GEMM
    xg_gemm_kernel<<<dim3((T_STEPS * BATCH) / 128, G4H / 128), 256, 0, stream>>>(
        Xb, WxB, bias, xg);

    // Phase 3: sequential recurrence, h ping-pong; 128KB dynamic LDS for h.
    for (int t = 0; t < T_STEPS; ++t) {
        const bf16_t* hin = (t & 1) ? h1 : h0;
        bf16_t*      hout = (t & 1) ? h0 : h1;
        lstm_step_kernel<<<HDIM / 32, 256, BATCH * HDIM * sizeof(bf16_t), stream>>>(
            xg, WhB, hin, hout, cbuf, out, t);
    }
}
